// MambaBlock_9552007266853
// MI455X (gfx1250) — matrix operand order repaired and checked by reading, not compiled, not run on hardware
//
#include <hip/hip_runtime.h>
#include <hip/hip_bf16.h>
#include <stddef.h>

#define B_SZ   2
#define DMODEL 256
#define DIN    512      // D_INNER
#define LTOK   8192     // 32*16*16
#define DST    16       // D_STATE
#define DTR    16       // DT_RANK
#define NCHUNK 64
#define CLEN   128      // LTOK / NCHUNK

typedef __bf16          v16bf __attribute__((ext_vector_type(16)));
typedef unsigned short  v16u  __attribute__((ext_vector_type(16)));
typedef float           v8f   __attribute__((ext_vector_type(8)));

__device__ __forceinline__ unsigned short f2bf(float f) {
    unsigned u = __builtin_bit_cast(unsigned, f);
    u += 0x7FFFu + ((u >> 16) & 1u);           // round-to-nearest-even
    return (unsigned short)(u >> 16);
}
__device__ __forceinline__ unsigned pk2bf(float a, float b) {
    return (unsigned)f2bf(a) | ((unsigned)f2bf(b) << 16);
}
__device__ __forceinline__ float silu_f(float x) {
    return x / (1.0f + __expf(-x));
}
__device__ __forceinline__ float softplus_f(float x) {
    return fmaxf(x, 0.0f) + __logf(1.0f + __expf(-fabsf(x)));
}
// branch sequence permutation: mode 0=f (identity), 1=b (flip), 2=s (slice interleave)
__device__ __forceinline__ int permf(int t, int mode) {
    if (mode == 0) return t;
    if (mode == 1) return (LTOK - 1) - t;
    return (t & 31) * 256 + (t >> 5);
}

// ---------------- LayerNorm:  x (B,C,L) -> xnorm (B,L,C)  ----------------
__global__ __launch_bounds__(256) void ln_kernel(const float* __restrict__ x,
                                                 const float* __restrict__ w,
                                                 const float* __restrict__ bias,
                                                 float* __restrict__ xn) {
    int l = blockIdx.x, b = blockIdx.y, c = threadIdx.x;
    float v = x[((size_t)b * DMODEL + c) * LTOK + l];
    __shared__ float s1[256], s2[256];
    s1[c] = v; s2[c] = v * v;
    __syncthreads();
    for (int o = 128; o > 0; o >>= 1) {
        if (c < o) { s1[c] += s1[c + o]; s2[c] += s2[c + o]; }
        __syncthreads();
    }
    float mu  = s1[0] * (1.0f / DMODEL);
    float var = s2[0] * (1.0f / DMODEL) - mu * mu;
    float rs  = rsqrtf(var + 1e-5f);
    xn[((size_t)b * LTOK + l) * DMODEL + c] = (v - mu) * rs * w[c] + bias[c];
}

// ---------------- bf16 WMMA GEMM:  C(M,N) = A(M,K) * W(N,K)^T ----------------
// A fp32 row-major, W fp32 row-major (N,K); converted to bf16 during LDS staging.
// LDS tiles are FRAGMENT-MAJOR [tile][lane][element]: each lane's 16-element
// WMMA fragment is one contiguous 32B vector load (2x ds_load_b128).
// Staging: for 4 consecutive 4-aligned k, both A and B mappings keep the same
// lane with consecutive e -> one float4 global load + one ds_store_b64 each.
#define BM 64
#define BN 128
#define BK 32
__global__ __launch_bounds__(256) void gemm_bf16_wmma(const float* __restrict__ A,
                                                      const float* __restrict__ W,
                                                      float* __restrict__ C,
                                                      int M, int N, int K) {
    // A: 4 m-tiles of 16x32 ; B: 8 n-tiles of 32x16
    __shared__ __align__(32) unsigned short As[4][32][16];   // 4 KB
    __shared__ __align__(32) unsigned short Bs[8][32][16];   // 8 KB
    int n0 = blockIdx.x * BN;
    int m0 = blockIdx.y * BM;
    int tid  = threadIdx.x;
    int wave = tid >> 5, lane = tid & 31;
    int mt = wave >> 1;            // 0..3 : wave's 16-row slab
    int nq = (wave & 1) * 4;       // wave covers n-tiles nq..nq+3 (64 cols)
    v8f acc[4] = {{}, {}, {}, {}};

    for (int k0 = 0; k0 < K; k0 += BK) {
        // stage A tile (64 rows x 32 k) : one float4 + one ds_store_b64 per group
        #pragma unroll
        for (int j = 0; j < (BM * BK) / (256 * 4); j++) {     // 2 iters
            int i  = tid + j * 256;
            int r  = i >> 3;                 // row in tile (0..63)
            int cg = (i & 7) * 4;            // k group start (0,4,...,28)
            const float4 v = *(const float4*)&A[(size_t)(m0 + r) * K + (k0 + cg)];
            uint2 p; p.x = pk2bf(v.x, v.y); p.y = pk2bf(v.z, v.w);
            int la = (r & 15) + (((cg >> 3) & 1) << 4);
            int e  = (cg & 7) + ((cg >> 4) << 3);
            *(uint2*)&As[r >> 4][la][e] = p;
        }
        // stage B tile (128 n x 32 k) from W (N,K) row-major
        #pragma unroll
        for (int j = 0; j < (BN * BK) / (256 * 4); j++) {     // 4 iters
            int i  = tid + j * 256;
            int n  = i >> 3;                 // n in tile (0..127)
            int cg = (i & 7) * 4;            // k group start
            float4 v = make_float4(0.f, 0.f, 0.f, 0.f);
            if (n0 + n < N) v = *(const float4*)&W[(size_t)(n0 + n) * K + (k0 + cg)];
            uint2 p; p.x = pk2bf(v.x, v.y); p.y = pk2bf(v.z, v.w);
            int lb = (n & 15) + (((cg >> 3) & 1) << 4);
            *(uint2*)&Bs[n >> 4][lb][(cg & 7) + ((cg >> 4) << 3)] = p;
        }
        __syncthreads();

        v16bf af = __builtin_bit_cast(v16bf, *(const v16u*)&As[mt][lane][0]);
        #pragma unroll
        for (int j = 0; j < 4; j++) {
            v16bf bfj = __builtin_bit_cast(v16bf, *(const v16u*)&Bs[nq + j][lane][0]);
            acc[j] = __builtin_amdgcn_wmma_f32_16x16x32_bf16(false, af, false, bfj,
                                                             (short)0, acc[j], false, false);
        }
        __syncthreads();
    }
    // C/D layout: row m = r + (lane>=16?8:0), col n = lane%16
    int mlo = (lane >> 4) << 3;
    #pragma unroll
    for (int j = 0; j < 4; j++) {
        int na = n0 + (nq + j) * 16 + (lane & 15);
        if (na < N) {
            #pragma unroll
            for (int r = 0; r < 8; r++) {
                int m = m0 + mt * 16 + r + mlo;
                C[(size_t)m * N + na] = acc[j][r];
            }
        }
    }
}

// ------------- depthwise causal conv(4) + SiLU, in branch-sequence order -------------
// xz (B,L,1024) ; xc (B,L,512) with xc[b][t][d] = silu(bias + sum_k w[d,k]*x[d, perm(t-3+k)])
__global__ __launch_bounds__(256) void conv_silu_kernel(const float* __restrict__ xz,
                                                        const float* __restrict__ cw,
                                                        const float* __restrict__ cb,
                                                        float* __restrict__ xc, int mode) {
    int t = blockIdx.x, b = blockIdx.y;
    for (int d = threadIdx.x; d < DIN; d += 256) {
        float acc = cb[d];
        #pragma unroll
        for (int k = 0; k < 4; k++) {
            int tk = t - 3 + k;
            if (tk >= 0) {
                int l = permf(tk, mode);
                acc += cw[d * 4 + k] * xz[((size_t)b * LTOK + l) * 1024 + d];
            }
        }
        xc[((size_t)b * LTOK + t) * DIN + d] = silu_f(acc);
    }
}

// ------------- selective-scan pass 1: per-chunk partial state (h_in = 0) -------------
// xdbl (B,L,48) = [dt(16) | B(16) | C(16)] per t. dt-projection folded in.
__global__ __launch_bounds__(256) void scan_pass1(const float* __restrict__ xdbl,
                                                  const float* __restrict__ xc,
                                                  const float* __restrict__ dtw,
                                                  const float* __restrict__ dtb,
                                                  const float* __restrict__ A_log,
                                                  float* __restrict__ Ssum,
                                                  float* __restrict__ hpart) {
    int d = blockIdx.x * 256 + threadIdx.x;
    int chunk = blockIdx.y, b = blockIdx.z;
    float An[DST], wr[DTR], h[DST];
    #pragma unroll
    for (int n = 0; n < DST; n++) { An[n] = -__expf(A_log[d * DST + n]); h[n] = 0.0f; }
    #pragma unroll
    for (int r = 0; r < DTR; r++) wr[r] = dtw[d * DTR + r];
    float bias = dtb[d], S = 0.0f;
    int t0 = chunk * CLEN;
    for (int tt = 0; tt < CLEN; tt++) {
        int t = t0 + tt;
        const float* xb = xdbl + ((size_t)b * LTOK + t) * 48;
        float xv = bias;
        #pragma unroll
        for (int r = 0; r < DTR; r++) xv += wr[r] * xb[r];
        float delta = softplus_f(xv);
        float u  = xc[((size_t)b * LTOK + t) * DIN + d];
        float du = delta * u;
        S += delta;
        #pragma unroll
        for (int n = 0; n < DST; n++)
            h[n] = __expf(delta * An[n]) * h[n] + du * xb[DTR + n];
    }
    size_t base = (size_t)(b * DIN + d) * NCHUNK + chunk;
    Ssum[base] = S;
    #pragma unroll
    for (int n = 0; n < DST; n++) hpart[base * DST + n] = h[n];
}

// ------------- pass 2: chunk-level combine (64 serial steps, tiny) -------------
__global__ __launch_bounds__(256) void scan_pass2(const float* __restrict__ A_log,
                                                  const float* __restrict__ Ssum,
                                                  const float* __restrict__ hpart,
                                                  float* __restrict__ hinit) {
    int idx = blockIdx.x * 256 + threadIdx.x;   // b*DIN + d
    int d = idx & (DIN - 1);
    float An[DST], h[DST];
    #pragma unroll
    for (int n = 0; n < DST; n++) { An[n] = -__expf(A_log[d * DST + n]); h[n] = 0.0f; }
    for (int c = 0; c < NCHUNK; c++) {
        size_t base = (size_t)idx * NCHUNK + c;
        #pragma unroll
        for (int n = 0; n < DST; n++) hinit[base * DST + n] = h[n];
        float S = Ssum[base];
        #pragma unroll
        for (int n = 0; n < DST; n++)
            h[n] = __expf(An[n] * S) * h[n] + hpart[base * DST + n];
    }
}

// ------------- pass 3: replay chunks from true initial state, emit gated y -------------
__global__ __launch_bounds__(256) void scan_pass3(const float* __restrict__ xdbl,
                                                  const float* __restrict__ xc,
                                                  const float* __restrict__ xz,
                                                  const float* __restrict__ dtw,
                                                  const float* __restrict__ dtb,
                                                  const float* __restrict__ A_log,
                                                  const float* __restrict__ Dv,
                                                  const float* __restrict__ hinit,
                                                  float* __restrict__ y,
                                                  int mode, int accum) {
    int d = blockIdx.x * 256 + threadIdx.x;
    int chunk = blockIdx.y, b = blockIdx.z;
    float An[DST], wr[DTR], h[DST];
    #pragma unroll
    for (int n = 0; n < DST; n++) An[n] = -__expf(A_log[d * DST + n]);
    #pragma unroll
    for (int r = 0; r < DTR; r++) wr[r] = dtw[d * DTR + r];
    size_t hb = ((size_t)(b * DIN + d) * NCHUNK + chunk) * DST;
    #pragma unroll
    for (int n = 0; n < DST; n++) h[n] = hinit[hb + n];
    float bias = dtb[d], Dd = Dv[d];
    int t0 = chunk * CLEN;
    for (int tt = 0; tt < CLEN; tt++) {
        int t = t0 + tt;
        const float* xb = xdbl + ((size_t)b * LTOK + t) * 48;
        float xv = bias;
        #pragma unroll
        for (int r = 0; r < DTR; r++) xv += wr[r] * xb[r];
        float delta = softplus_f(xv);
        float u  = xc[((size_t)b * LTOK + t) * DIN + d];
        float du = delta * u;
        float acc = 0.0f;
        #pragma unroll
        for (int n = 0; n < DST; n++) {
            h[n] = __expf(delta * An[n]) * h[n] + du * xb[DTR + n];
            acc += h[n] * xb[2 * DTR + n];
        }
        acc += Dd * u;
        int l = permf(t, mode);
        float z = xz[((size_t)b * LTOK + l) * 1024 + DIN + d];
        acc *= silu_f(z);
        size_t addr = ((size_t)b * LTOK + l) * DIN + d;
        if (accum) y[addr] += acc; else y[addr] = acc;
    }
}

// ------------- final transpose: (B,L,256) -> (B,256,L) = (B,C,32,16,16) -------------
__global__ void transpose_out(const float* __restrict__ in, float* __restrict__ out) {
    __shared__ float tile[32][33];
    int b = blockIdx.z;
    int c0 = blockIdx.x * 32, l0 = blockIdx.y * 32;
    int tx = threadIdx.x, ty = threadIdx.y;       // 32 x 8
    for (int i = 0; i < 32; i += 8)
        tile[ty + i][tx] = in[((size_t)b * LTOK + l0 + ty + i) * DMODEL + c0 + tx];
    __syncthreads();
    for (int i = 0; i < 32; i += 8)
        out[((size_t)b * DMODEL + c0 + ty + i) * LTOK + l0 + tx] = tile[tx][ty + i];
}

extern "C" void kernel_launch(void* const* d_in, const int* in_sizes, int n_in,
                              void* d_out, int out_size, void* d_ws, size_t ws_size,
                              hipStream_t stream) {
    (void)in_sizes; (void)n_in; (void)out_size; (void)ws_size;
    const float* x         = (const float*)d_in[0];
    const float* ln_w      = (const float*)d_in[1];
    const float* ln_b      = (const float*)d_in[2];
    const float* in_proj_w = (const float*)d_in[3];
    const float* out_proj_w= (const float*)d_in[4];

    // workspace carve-up (floats)
    float* ws = (float*)d_ws;
    float* xnorm  = ws;                           // B*L*256
    float* xz     = xnorm  + (size_t)B_SZ * LTOK * DMODEL;   // B*L*1024
    float* xc     = xz     + (size_t)B_SZ * LTOK * 1024;     // B*L*512
    float* xdbl   = xc     + (size_t)B_SZ * LTOK * DIN;      // B*L*48
    float* Ssum   = xdbl   + (size_t)B_SZ * LTOK * 48;       // B*512*64
    float* hpart  = Ssum   + (size_t)B_SZ * DIN * NCHUNK;    // B*512*64*16
    float* hinit  = hpart  + (size_t)B_SZ * DIN * NCHUNK * DST;
    float* ybuf   = hinit  + (size_t)B_SZ * DIN * NCHUNK * DST;  // B*L*512
    float* outtmp = ybuf   + (size_t)B_SZ * LTOK * DIN;          // B*L*256

    const int M = B_SZ * LTOK;   // 16384 rows across both batches (contiguous)

    // 1) LayerNorm + transpose to (B,L,C)
    ln_kernel<<<dim3(LTOK, B_SZ), 256, 0, stream>>>(x, ln_w, ln_b, xnorm);

    // 2) in_proj GEMM: (M,256) x (1024,256)^T -> xz (M,1024)
    gemm_bf16_wmma<<<dim3(1024 / BN, M / BM), 256, 0, stream>>>(xnorm, in_proj_w, xz,
                                                                M, 1024, DMODEL);

    // 3) branches f, b, s (mode = branch index)
    for (int br = 0; br < 3; br++) {
        const float* cw  = (const float*)d_in[5 + 7 * br + 0];
        const float* cb  = (const float*)d_in[5 + 7 * br + 1];
        const float* xpw = (const float*)d_in[5 + 7 * br + 2];
        const float* dtw = (const float*)d_in[5 + 7 * br + 3];
        const float* dtb = (const float*)d_in[5 + 7 * br + 4];
        const float* Al  = (const float*)d_in[5 + 7 * br + 5];
        const float* Dv  = (const float*)d_in[5 + 7 * br + 6];

        conv_silu_kernel<<<dim3(LTOK, B_SZ), 256, 0, stream>>>(xz, cw, cb, xc, br);

        // x-projection: (M,512) x (48,512)^T -> xdbl (M,48)
        gemm_bf16_wmma<<<dim3(1, M / BM), 256, 0, stream>>>(xc, xpw, xdbl, M, 48, DIN);

        scan_pass1<<<dim3(DIN / 256, NCHUNK, B_SZ), 256, 0, stream>>>(xdbl, xc, dtw, dtb,
                                                                      Al, Ssum, hpart);
        scan_pass2<<<dim3((B_SZ * DIN) / 256), 256, 0, stream>>>(Al, Ssum, hpart, hinit);
        scan_pass3<<<dim3(DIN / 256, NCHUNK, B_SZ), 256, 0, stream>>>(xdbl, xc, xz, dtw,
                                                                      dtb, Al, Dv, hinit,
                                                                      ybuf, br, br > 0);
    }

    // 4) out_proj GEMM: (M,512) x (256,512)^T -> outtmp (M,256)
    gemm_bf16_wmma<<<dim3(DMODEL / BN, M / BM), 256, 0, stream>>>(ybuf, out_proj_w, outtmp,
                                                                  M, DMODEL, DIN);

    // 5) transpose back to (B,C,32,16,16)
    transpose_out<<<dim3(DMODEL / 32, LTOK / 32, B_SZ), dim3(32, 8), 0, stream>>>(
        outtmp, (float*)d_out);
}
